// MHA_56693568307397
// MI455X (gfx1250) — matrix operand order repaired and checked by reading, not compiled, not run on hardware
//
#include <hip/hip_runtime.h>
#include <hip/hip_bf16.h>

// Problem constants (reference: B=2, S=2048, D=1024, H=16, HD=64)
#define BB 2
#define SS 2048
#define DD 1024
#define HH 16
#define HDIM 64
#define BSROWS (BB * SS)   // 4096

typedef __bf16 bf16;
typedef __attribute__((ext_vector_type(16))) __bf16 v16bf;
typedef __attribute__((ext_vector_type(8)))  float  v8f;

__device__ inline v8f v8f_zero() {
    v8f z = {0.f, 0.f, 0.f, 0.f, 0.f, 0.f, 0.f, 0.f};
    return z;
}

// Load a 16-element bf16 A/B fragment: two 16-byte chunks from p0 and p1.
// Elements 0..7 of the v16bf come from p0, elements 8..15 from p1.
__device__ inline v16bf load_frag2(const bf16* p0, const bf16* p1) {
    union { uint4 u[2]; v16bf v; } t;
    t.u[0] = *reinterpret_cast<const uint4*>(p0);
    t.u[1] = *reinterpret_cast<const uint4*>(p1);
    return t.v;
}

__device__ inline v8f wmma_bf16(v16bf a, v16bf b, v8f c) {
    // D = A(16x32 bf16) * B(32x16 bf16) + C(16x16 f32)
    return __builtin_amdgcn_wmma_f32_16x16x32_bf16(
        /*neg_a=*/false, a, /*neg_b=*/false, b,
        /*c_mod=*/(short)0, c, /*reuse_a=*/false, /*reuse_b=*/false);
}

// Async global -> LDS copy, 16B per lane, tracked by ASYNCcnt.
// lds_off: LDS byte address (per lane), voff: 32-bit byte offset (per lane),
// gbase: uniform 64-bit global base (SGPR pair).
__device__ inline void async_g2l_b128(unsigned lds_off, unsigned voff,
                                      unsigned long long gbase) {
    asm volatile("global_load_async_to_lds_b128 %0, %1, %2"
                 :: "v"(lds_off), "v"(voff), "s"(gbase)
                 : "memory");
}

#define S_WAIT_ASYNCCNT(n) asm volatile("s_wait_asynccnt " #n ::: "memory")

// ---------------------------------------------------------------------------
// Kernel 0: fp32 -> bf16 conversion (x and the four weight matrices)
// ---------------------------------------------------------------------------
__global__ __launch_bounds__(256) void cvt_bf16_kernel(const float* __restrict__ in,
                                                       bf16* __restrict__ out, int n) {
    int i = (blockIdx.x * blockDim.x + threadIdx.x) * 4;
    if (i + 3 < n) {
        float4 v = *reinterpret_cast<const float4*>(in + i);
        out[i + 0] = (bf16)v.x;
        out[i + 1] = (bf16)v.y;
        out[i + 2] = (bf16)v.z;
        out[i + 3] = (bf16)v.w;
    }
}

// ---------------------------------------------------------------------------
// Kernel 1: fused QKV projection (+bias) + RoPE on Q,K + V transpose.
// One wave computes a 32(M) x 64(N) tile of x[BS,D] @ W^T : two A-fragment
// rows share every B-fragment (8 WMMA per 6KB of fragment traffic).
//   blockIdx.y: 0 -> Q, 1 -> K, 2 -> V
// Outputs:
//   Qr, Kr : bf16 [b, h, s, hd]   (RoPE applied)
//   Vt     : bf16 [b, h, hd, s]   (transposed so P*V B-fragments are contiguous)
// ---------------------------------------------------------------------------
__global__ __launch_bounds__(128) void qkv_kernel(
    const bf16* __restrict__ X,
    const bf16* __restrict__ Wq, const bf16* __restrict__ Wk, const bf16* __restrict__ Wv,
    const float* __restrict__ bq, const float* __restrict__ bk, const float* __restrict__ bv,
    const float* __restrict__ fcos, const float* __restrict__ fsin,
    bf16* __restrict__ Qr, bf16* __restrict__ Kr, bf16* __restrict__ Vt)
{
    const int lane  = threadIdx.x & 31;
    const int wid   = threadIdx.x >> 5;
    const int w     = blockIdx.y;                 // which projection
    const int tile  = blockIdx.x * 4 + wid;       // [0, 2048)
    const int m0    = (tile >> 4) * 32;           // row block in [0, BS)
    const int n0    = (tile & 15) * 64;           // col block in [0, D)

    const bf16*  W    = (w == 0) ? Wq : (w == 1) ? Wk : Wv;
    const float* bias = (w == 0) ? bq : (w == 1) ? bk : bv;

    const int col0    = lane & 15;
    const int khalf   = lane >> 4;
    const int rowbase = khalf * 8;

    const bf16* arow0 = X + (size_t)(m0 + col0) * DD;
    const bf16* arow1 = X + (size_t)(m0 + 16 + col0) * DD;

    v8f acc[8];
#pragma unroll
    for (int j = 0; j < 8; ++j) acc[j] = v8f_zero();

    for (int k0 = 0; k0 < DD; k0 += 32) {
        v16bf a0 = load_frag2(arow0 + k0 + 16 * khalf, arow0 + k0 + 8 + 16 * khalf);
        v16bf a1 = load_frag2(arow1 + k0 + 16 * khalf, arow1 + k0 + 8 + 16 * khalf);
        __builtin_prefetch(arow0 + k0 + 64, 0, 1);
#pragma unroll
        for (int j = 0; j < 4; ++j) {
            const bf16* brow = W + (size_t)(n0 + j * 16 + col0) * DD + k0 + 16 * khalf;
            v16bf b = load_frag2(brow, brow + 8);
            acc[j]     = wmma_bf16(a0, b, acc[j]);
            acc[4 + j] = wmma_bf16(a1, b, acc[4 + j]);
        }
    }

    if (w == 2) {
        // V: store transposed [b, h, hd, s]
#pragma unroll
        for (int j = 0; j < 4; ++j) {
            const int col = n0 + j * 16 + col0;
            const float bias_v = bias[col];
            const int h = col >> 6, hd = col & 63;
#pragma unroll
            for (int blk = 0; blk < 2; ++blk) {
#pragma unroll
                for (int r = 0; r < 8; ++r) {
                    const int row = m0 + blk * 16 + rowbase + r;
                    const int b_ = row >> 11, s = row & (SS - 1);
                    const float v = acc[blk * 4 + j][r] + bias_v;
                    Vt[((size_t)(b_ * HH + h) * HDIM + hd) * SS + s] = (bf16)v;
                }
            }
        }
    } else {
        // Q/K: RoPE then store [b, h, s, hd]
        bf16* Out = (w == 0) ? Qr : Kr;
#pragma unroll
        for (int j = 0; j < 4; ++j) {
            const int col = n0 + j * 16 + col0;
            const float bias_v = bias[col];
            const int h = col >> 6, hd = col & 63;
            const int fi = hd >> 1;
            const bool even = (hd & 1) == 0;
#pragma unroll
            for (int blk = 0; blk < 2; ++blk) {
#pragma unroll
                for (int r = 0; r < 8; ++r) {
                    const int row = m0 + blk * 16 + rowbase + r;
                    const int b_ = row >> 11, s = row & (SS - 1);
                    const float v = acc[blk * 4 + j][r] + bias_v;
                    // RoPE pair (hd even, hd odd) sit in adjacent lanes (N dim).
                    const float p = __shfl_xor(v, 1, 32);
                    const float c  = fcos[s * (HDIM / 2) + fi];
                    const float sn = fsin[s * (HDIM / 2) + fi];
                    const float y = even ? (v * c - p * sn) : (p * sn + v * c);
                    Out[((size_t)(b_ * HH + h) * SS + s) * HDIM + hd] = (bf16)y;
                }
            }
        }
    }
}

// ---------------------------------------------------------------------------
// Kernel 2: causal flash attention with async double-buffered K/V staging.
// Block = 128 threads (4 waves) = 64 consecutive queries of one (b, h).
// Each 32-key chunk of K (32x64) and V^T (64x32) is streamed into LDS once
// per block with global_load_async_to_lds_b128 and shared by all 4 waves.
// Chunk c+1 streams while chunk c computes (s_wait_asynccnt 4 pipelining).
// ---------------------------------------------------------------------------
__global__ __launch_bounds__(128) void attn_kernel(
    const bf16* __restrict__ Qr, const bf16* __restrict__ Kr,
    const bf16* __restrict__ Vt, bf16* __restrict__ Obf)
{
    // LDS layout (bytes):
    //   [0,     8192)  K bufs: 2 x (32 keys x 64 hd)   bf16
    //   [8192, 16384)  V bufs: 2 x (64 hd  x 32 keys)  bf16
    //   [16384,20480)  P: 4 waves x (16 q x 32 keys)   bf16
    __shared__ bf16 smem[10240];

    const int tid     = threadIdx.x;
    const int lane    = tid & 31;
    const int wid     = tid >> 5;
    const int col0    = lane & 15;
    const int khalf   = lane >> 4;
    const int rowbase = khalf * 8;

    const int qb = blockIdx.x & 31;            // 64-query block
    const int h  = (blockIdx.x >> 5) & 15;
    const int b  = blockIdx.x >> 9;
    const int q0b = qb * 64;
    const int q0  = q0b + wid * 16;            // this wave's query tile

    const bf16* Qbase = Qr + (size_t)(b * HH + h) * SS * HDIM;
    const unsigned long long KbaseB =
        (unsigned long long)(size_t)(Kr + (size_t)(b * HH + h) * SS * HDIM);
    const unsigned long long VbaseB =
        (unsigned long long)(size_t)(Vt + (size_t)(b * HH + h) * HDIM * SS);

    const unsigned smem_base = (unsigned)(size_t)(&smem[0]);  // LDS addr = addr[31:0]

    // Per-thread async copy assignments (2 x 16B for K, 2 x 16B for V).
    // K chunk: 32 rows x 128 B;  V chunk: 64 rows x 64 B (row stride in mem: 4096 B).
    const int ik0 = tid, ik1 = tid + 128;
    const unsigned voffK0 = (unsigned)((ik0 >> 3) * 128 + (ik0 & 7) * 16);
    const unsigned voffK1 = (unsigned)((ik1 >> 3) * 128 + (ik1 & 7) * 16);
    const unsigned ldsK0  = smem_base + voffK0;
    const unsigned ldsK1  = smem_base + voffK1;
    const unsigned voffV0 = (unsigned)((ik0 >> 2) * 4096 + (ik0 & 3) * 16);
    const unsigned voffV1 = (unsigned)((ik1 >> 2) * 4096 + (ik1 & 3) * 16);
    const unsigned ldsV0  = smem_base + 8192u + (unsigned)((ik0 >> 2) * 64 + (ik0 & 3) * 16);
    const unsigned ldsV1  = smem_base + 8192u + (unsigned)((ik1 >> 2) * 64 + (ik1 & 3) * 16);

    auto issue_chunk = [&](int c) {
        const unsigned p = (unsigned)(c & 1) * 4096u;
        const unsigned long long gK = KbaseB + (unsigned long long)c * 4096ull;
        const unsigned long long gV = VbaseB + (unsigned long long)c * 64ull;
        async_g2l_b128(ldsK0 + p, voffK0, gK);
        async_g2l_b128(ldsK1 + p, voffK1, gK);
        async_g2l_b128(ldsV0 + p, voffV0, gV);
        async_g2l_b128(ldsV1 + p, voffV1, gV);
    };

    // Q A-fragments: HD=64 -> two K-steps of 32, kept resident.
    const bf16* qrow = Qbase + (size_t)(q0 + col0) * HDIM;
    const v16bf aQ0 = load_frag2(qrow +  0 + 16 * khalf, qrow +  8 + 16 * khalf);
    const v16bf aQ1 = load_frag2(qrow + 32 + 16 * khalf, qrow + 40 + 16 * khalf);

    float m[8], l[8];
    v8f o[4];
#pragma unroll
    for (int r = 0; r < 8; ++r) { m[r] = -1e30f; l[r] = 0.f; }
#pragma unroll
    for (int j = 0; j < 4; ++j) o[j] = v8f_zero();

    bf16* pw = smem + 8192 + wid * 512;        // this wave's P tile (elements)

    const int nchunks = qb * 2 + 2;            // keys up to block diagonal
    issue_chunk(0);

    for (int c = 0; c < nchunks; ++c) {
        __syncthreads();   // prior readers of buffer (c+1)&1 are done
        if (c + 1 < nchunks) {
            issue_chunk(c + 1);
            S_WAIT_ASYNCCNT(0x4);   // chunk c complete; c+1 still in flight
        } else {
            S_WAIT_ASYNCCNT(0x0);
        }
        __syncthreads();   // chunk c visible to all waves

        const bf16* kp = smem + (size_t)((c & 1) * 2048);          // 32 x 64
        const bf16* vp = smem + 4096 + (size_t)((c & 1) * 2048);   // 64 x 32
        const int kb = c * 32;

        // ---- S = Q K^T for 32 keys (two 16-key tiles) ----
        v8f s0 = v8f_zero(), s1 = v8f_zero();
        {
            const bf16* kr0 = kp + (size_t)col0 * 64 + 16 * khalf;
            s0 = wmma_bf16(aQ0, load_frag2(kr0,      kr0 + 8),  s0);
            s0 = wmma_bf16(aQ1, load_frag2(kr0 + 32, kr0 + 40), s0);
            const bf16* kr1 = kp + (size_t)(16 + col0) * 64 + 16 * khalf;
            s1 = wmma_bf16(aQ0, load_frag2(kr1,      kr1 + 8),  s1);
            s1 = wmma_bf16(aQ1, load_frag2(kr1 + 32, kr1 + 40), s1);
        }

        // ---- online softmax (row stats align with accumulator layout) ----
#pragma unroll
        for (int r = 0; r < 8; ++r) {
            const int q    = q0 + rowbase + r;
            const int key0 = kb + col0;
            const int key1 = kb + 16 + col0;
            float v0 = s0[r] * 0.125f;                  // 1/sqrt(64)
            float v1 = s1[r] * 0.125f;
            if (key0 > q) v0 = -1e9f;
            if (key1 > q) v1 = -1e9f;

            float rm = fmaxf(v0, v1);
            rm = fmaxf(rm, __shfl_xor(rm, 1, 32));
            rm = fmaxf(rm, __shfl_xor(rm, 2, 32));
            rm = fmaxf(rm, __shfl_xor(rm, 4, 32));
            rm = fmaxf(rm, __shfl_xor(rm, 8, 32));

            const float mn    = fmaxf(m[r], rm);
            const float alpha = __expf(m[r] - mn);
            const float p0    = __expf(v0 - mn);
            const float p1    = __expf(v1 - mn);
            float rs = p0 + p1;
            rs += __shfl_xor(rs, 1, 32);
            rs += __shfl_xor(rs, 2, 32);
            rs += __shfl_xor(rs, 4, 32);
            rs += __shfl_xor(rs, 8, 32);

            l[r] = l[r] * alpha + rs;
            m[r] = mn;
            o[0][r] *= alpha; o[1][r] *= alpha;
            o[2][r] *= alpha; o[3][r] *= alpha;

            pw[(rowbase + r) * 32 + col0]      = (bf16)p0;
            pw[(rowbase + r) * 32 + 16 + col0] = (bf16)p1;
        }
        __syncthreads();   // P tiles ready (C-layout -> A-layout via LDS)

        // P A-fragment (16 queries x 32 keys)
        const v16bf pA = load_frag2(pw + col0 * 32 + 16 * khalf,
                                    pw + col0 * 32 + 8 + 16 * khalf);

        // ---- O += P * V (V staged as [hd, key], K-dim contiguous) ----
#pragma unroll
        for (int j = 0; j < 4; ++j) {
            const bf16* vr = vp + (size_t)(j * 16 + col0) * 32 + 16 * khalf;
            o[j] = wmma_bf16(pA, load_frag2(vr, vr + 8), o[j]);
        }
    }

    // ---- normalize and store O as [b, s, h*HD + hd] (bf16) ----
#pragma unroll
    for (int j = 0; j < 4; ++j) {
        const int colh = h * HDIM + j * 16 + col0;
#pragma unroll
        for (int r = 0; r < 8; ++r) {
            const int q = q0 + rowbase + r;
            const float v = o[j][r] / l[r];
            Obf[(size_t)(b * SS + q) * DD + colh] = (bf16)v;
        }
    }
}

// ---------------------------------------------------------------------------
// Kernel 3: output projection  out = O @ Wo^T + bo  (fp32 result), 32x64 tiles
// ---------------------------------------------------------------------------
__global__ __launch_bounds__(128) void oproj_kernel(
    const bf16* __restrict__ A, const bf16* __restrict__ Wo,
    const float* __restrict__ bo, float* __restrict__ Out)
{
    const int lane    = threadIdx.x & 31;
    const int wid     = threadIdx.x >> 5;
    const int tile    = blockIdx.x * 4 + wid;     // [0, 2048)
    const int m0      = (tile >> 4) * 32;
    const int n0      = (tile & 15) * 64;
    const int col0    = lane & 15;
    const int khalf   = lane >> 4;
    const int rowbase = khalf * 8;

    const bf16* arow0 = A + (size_t)(m0 + col0) * DD;
    const bf16* arow1 = A + (size_t)(m0 + 16 + col0) * DD;

    v8f acc[8];
#pragma unroll
    for (int j = 0; j < 8; ++j) acc[j] = v8f_zero();

    for (int k0 = 0; k0 < DD; k0 += 32) {
        v16bf a0 = load_frag2(arow0 + k0 + 16 * khalf, arow0 + k0 + 8 + 16 * khalf);
        v16bf a1 = load_frag2(arow1 + k0 + 16 * khalf, arow1 + k0 + 8 + 16 * khalf);
        __builtin_prefetch(arow0 + k0 + 64, 0, 1);
#pragma unroll
        for (int j = 0; j < 4; ++j) {
            const bf16* brow = Wo + (size_t)(n0 + j * 16 + col0) * DD + k0 + 16 * khalf;
            v16bf b = load_frag2(brow, brow + 8);
            acc[j]     = wmma_bf16(a0, b, acc[j]);
            acc[4 + j] = wmma_bf16(a1, b, acc[4 + j]);
        }
    }

#pragma unroll
    for (int j = 0; j < 4; ++j) {
        const int col = n0 + j * 16 + col0;
        const float bias_v = bo[col];
#pragma unroll
        for (int blk = 0; blk < 2; ++blk) {
#pragma unroll
            for (int r = 0; r < 8; ++r) {
                const int row = m0 + blk * 16 + rowbase + r;
                Out[(size_t)row * DD + col] = acc[blk * 4 + j][r] + bias_v;
            }
        }
    }
}

// ---------------------------------------------------------------------------
// Host launcher
// ---------------------------------------------------------------------------
extern "C" void kernel_launch(void* const* d_in, const int* in_sizes, int n_in,
                              void* d_out, int out_size, void* d_ws, size_t ws_size,
                              hipStream_t stream) {
    const float* x    = (const float*)d_in[0];
    const float* wq   = (const float*)d_in[1];
    const float* bq   = (const float*)d_in[2];
    const float* wk   = (const float*)d_in[3];
    const float* bk   = (const float*)d_in[4];
    const float* wv   = (const float*)d_in[5];
    const float* bv   = (const float*)d_in[6];
    const float* wo   = (const float*)d_in[7];
    const float* bo   = (const float*)d_in[8];
    const float* fcos = (const float*)d_in[9];
    const float* fsin = (const float*)d_in[10];
    // d_in[11] = mask: causal triu, computed analytically in-kernel.
    float* out = (float*)d_out;

    char* ws = (char*)d_ws;
    bf16* Xbf = (bf16*)(ws + 0);          //  8,388,608 B
    bf16* Wqb = (bf16*)(ws + 8388608);    //  2,097,152 B
    bf16* Wkb = (bf16*)(ws + 10485760);
    bf16* Wvb = (bf16*)(ws + 12582912);
    bf16* Wob = (bf16*)(ws + 14680064);
    bf16* Qr  = (bf16*)(ws + 16777216);   //  8,388,608 B  [b,h,s,hd]
    bf16* Kr  = (bf16*)(ws + 25165824);   //  8,388,608 B  [b,h,s,hd]
    bf16* Vt  = (bf16*)(ws + 33554432);   //  8,388,608 B  [b,h,hd,s]
    bf16* Obf = (bf16*)(ws + 41943040);   //  8,388,608 B  [b,s,d]

    const int nX = BB * SS * DD;   // 4,194,304
    const int nW = DD * DD;        // 1,048,576

    cvt_bf16_kernel<<<nX / (256 * 4), 256, 0, stream>>>(x,  Xbf, nX);
    cvt_bf16_kernel<<<nW / (256 * 4), 256, 0, stream>>>(wq, Wqb, nW);
    cvt_bf16_kernel<<<nW / (256 * 4), 256, 0, stream>>>(wk, Wkb, nW);
    cvt_bf16_kernel<<<nW / (256 * 4), 256, 0, stream>>>(wv, Wvb, nW);
    cvt_bf16_kernel<<<nW / (256 * 4), 256, 0, stream>>>(wo, Wob, nW);

    // 2048 32x64 tiles per projection, 4 waves per block, 3 projections.
    qkv_kernel<<<dim3(512, 3), 128, 0, stream>>>(Xbf, Wqb, Wkb, Wvb,
                                                 bq, bk, bv, fcos, fsin,
                                                 Qr, Kr, Vt);

    // 4 waves (64 queries) per block per (b, h): 2*16*32 = 1024 blocks.
    attn_kernel<<<1024, 128, 0, stream>>>(Qr, Kr, Vt, Obf);

    oproj_kernel<<<512, 128, 0, stream>>>(Obf, Wob, bo, out);
}
